// ODEJumpFunc_27195732918844
// MI455X (gfx1250) — matrix operand order repaired and checked by reading, not compiled, not run on hardware
//
#include <hip/hip_runtime.h>

// ---------------------------------------------------------------------------
// Problem constants (from the reference)
// ---------------------------------------------------------------------------
#define S_    4
#define N_    50000
#define E_    800000
#define DIMC  8
#define DIMH  8
#define HID   32
#define DZ    16          // DIM_C + DIM_H

typedef __attribute__((ext_vector_type(16))) __bf16 v16bf;
typedef __attribute__((ext_vector_type(8)))  float  v8f;

// Branchless fast transcendentals (v_exp_f32 / v_log_f32 are 1-cycle TRANS ops
// on CDNA5; the libm versions expand to range-checked exec-mask diamonds).
__device__ __forceinline__ float celu1(float x) {
    // max(x,0) + min(exp(x)-1, 0); exp overflow->inf gives min(inf,0)=0: safe.
    return fmaxf(x, 0.0f) + fminf(__expf(x) - 1.0f, 0.0f);
}
__device__ __forceinline__ float softplus1(float x) {
    // max(x,0) + log(1 + exp(-|x|)) : branchless, saturates correctly.
    return fmaxf(x, 0.0f) + __logf(1.0f + __expf(-fabsf(x)));
}
__device__ __forceinline__ float tanh_fast(float x) {
    // tanh(x) = sign(x) * (1 - 2/(exp(2|x|)+1)); exp->inf gives exactly 1.
    const float t = 1.0f - 2.0f / (__expf(2.0f * fabsf(x)) + 1.0f);
    return __builtin_copysignf(t, x);
}

// ---------------------------------------------------------------------------
// WMMA operand loaders (CDNA5 wave32 layouts, ISA 7.12.2)
// 16-bit A matrix 16x32 (MxK):
//   lane<16 : M=lane,    vgpr v<4 -> K=2v,2v+1 ; v>=4 -> K=16+2(v-4),...
//   lane>=16: M=lane-16, same K pattern shifted by +8
// ---------------------------------------------------------------------------
__device__ __forceinline__ v16bf load_A_lds(const float* tile /*16x32*/, int lane) {
    const int grp = lane >> 4;
    const int M   = lane & 15;
    v16bf a;
#pragma unroll
    for (int v = 0; v < 8; ++v) {
        const int kbase = (v < 4) ? (2 * v + 8 * grp)
                                  : (16 + 2 * (v - 4) + 8 * grp);
        a[2 * v + 0] = (__bf16)tile[M * 32 + kbase + 0];
        a[2 * v + 1] = (__bf16)tile[M * 32 + kbase + 1];
    }
    return a;
}

// 16-bit B matrix 32x16 (KxN): vgpr v, lane<16: N=lane, v<4 -> K=2v,2v+1; v>=4 -> K=16+2(v-4),...
// lane>=16: N=lane-16, same K pattern shifted by +8.  W row-major (fan_in x 32); k>=kmax -> 0.
__device__ __forceinline__ v16bf load_B(const float* __restrict__ W, int ldw,
                                        int kmax, int block, int lane) {
    const int grp = lane >> 4;
    const int n   = block * 16 + (lane & 15);
    v16bf bm;
#pragma unroll
    for (int v = 0; v < 8; ++v) {
#pragma unroll
        for (int h = 0; h < 2; ++h) {
            const int k = ((v < 4) ? (2 * v) : (16 + 2 * (v - 4))) + h + 8 * grp;
            const float w = (k < kmax) ? W[k * ldw + n] : 0.0f;
            bm[2 * v + h] = (__bf16)w;
        }
    }
    return bm;
}

// ---------------------------------------------------------------------------
// One 2-layer MLP (DZ->HID->HID, celu after both layers) for a 16-node tile,
// with B matrices and per-lane biases already resident in registers.
// ---------------------------------------------------------------------------
__device__ __forceinline__ void mlp2_tile(const float* zt, float* act, int lane,
                                          const v16bf* B1, const float* b1v,
                                          const v16bf* B2, const float* b2v,
                                          float* __restrict__ out,
                                          size_t outBase /* tile_node0 * HID */) {
    const int grp  = lane >> 4;
    const int nloc = lane & 15;

    // layer 1: z(16x16 padded to K=32) @ W1 -> 16x32, celu, to LDS
    const v16bf a1 = load_A_lds(zt, lane);
#pragma unroll
    for (int b = 0; b < 2; ++b) {
        v8f acc;
#pragma unroll
        for (int j = 0; j < 8; ++j) acc[j] = b1v[b];
        acc = __builtin_amdgcn_wmma_f32_16x16x32_bf16(
            false, a1, false, B1[b], (short)0, acc, false, false);
#pragma unroll
        for (int j = 0; j < 8; ++j)
            act[(j + 8 * grp) * 32 + b * 16 + nloc] = celu1(acc[j]);
    }
    __builtin_amdgcn_wave_barrier();   // order LDS writes vs. cross-lane reads

    // layer 2: act(16x32) @ W2 -> 16x32, celu, to global
    const v16bf a2 = load_A_lds(act, lane);
#pragma unroll
    for (int b = 0; b < 2; ++b) {
        v8f acc;
#pragma unroll
        for (int j = 0; j < 8; ++j) acc[j] = b2v[b];
        acc = __builtin_amdgcn_wmma_f32_16x16x32_bf16(
            false, a2, false, B2[b], (short)0, acc, false, false);
#pragma unroll
        for (int j = 0; j < 8; ++j)
            out[outBase + (size_t)(j + 8 * grp) * HID + b * 16 + nloc] = celu1(acc[j]);
    }
    __builtin_amdgcn_wave_barrier();
}

// ---------------------------------------------------------------------------
// Kernel 1: vcur / vnbr MLPs via WMMA. Persistent waves: weights converted to
// bf16 B-layout ONCE per wave (8 x v16bf in VGPRs), then grid-stride tiles.
// ---------------------------------------------------------------------------
#define MLP_BLOCKS 256

__global__ void __launch_bounds__(256)
mlp_kernel(const float* __restrict__ z,
           const float* __restrict__ cW1, const float* __restrict__ cb1,
           const float* __restrict__ cW2, const float* __restrict__ cb2,
           const float* __restrict__ nW1, const float* __restrict__ nb1,
           const float* __restrict__ nW2, const float* __restrict__ nb2,
           float* __restrict__ vcur, float* __restrict__ vnbr, int ntiles) {
    __shared__ float ztile[8][16 * 32];
    __shared__ float act[8][16 * 32];

    const int wave   = threadIdx.x >> 5;
    const int lane   = threadIdx.x & 31;
    const int wid    = blockIdx.x * 8 + wave;
    const int nwaves = MLP_BLOCKS * 8;
    const int nloc   = lane & 15;

    // ---- hoist all weight operands into registers (reused across tiles)
    v16bf cB1[2], cB2[2], nB1[2], nB2[2];
    float cb1v[2], cb2v[2], nb1v[2], nb2v[2];
#pragma unroll
    for (int b = 0; b < 2; ++b) {
        cB1[b] = load_B(cW1, HID, DZ,  b, lane);  cb1v[b] = cb1[b * 16 + nloc];
        cB2[b] = load_B(cW2, HID, HID, b, lane);  cb2v[b] = cb2[b * 16 + nloc];
        nB1[b] = load_B(nW1, HID, DZ,  b, lane);  nb1v[b] = nb1[b * 16 + nloc];
        nB2[b] = load_B(nW2, HID, HID, b, lane);  nb2v[b] = nb2[b * 16 + nloc];
    }

    float* zt = ztile[wave];
    const int row = lane >> 1;            // each lane stages half a z row
    const int c8  = (lane & 1) * 8;

    for (int tile = wid; tile < ntiles; tile += nwaves) {
        const size_t node0 = (size_t)tile * 16;

        // prefetch next tile's z rows (global_prefetch_b8)
        if (tile + nwaves < ntiles)
            __builtin_prefetch(z + ((size_t)(tile + nwaves) * 16 + row) * DZ, 0, 1);

        // stage z tile (16 nodes x 16 feats) into LDS, pad K 16..31 with zeros
        {
            const float4* zr =
                reinterpret_cast<const float4*>(z + (node0 + row) * DZ + c8);
            const float4 v0 = zr[0];
            const float4 v1 = zr[1];
            float* d = zt + row * 32 + c8;
            d[0] = v0.x; d[1] = v0.y; d[2] = v0.z; d[3] = v0.w;
            d[4] = v1.x; d[5] = v1.y; d[6] = v1.z; d[7] = v1.w;
            float* dz = zt + row * 32 + 16 + c8;
            dz[0] = 0.f; dz[1] = 0.f; dz[2] = 0.f; dz[3] = 0.f;
            dz[4] = 0.f; dz[5] = 0.f; dz[6] = 0.f; dz[7] = 0.f;
        }
        __builtin_amdgcn_wave_barrier();

        const size_t outBase = node0 * HID;
        mlp2_tile(zt, act[wave], lane, cB1, cb1v, cB2, cb2v, vcur, outBase);
        mlp2_tile(zt, act[wave], lane, nB1, nb1v, nB2, nb2v, vnbr, outBase);
    }
}

// ---------------------------------------------------------------------------
// Kernel 2: edge scatter-add. 32 lanes per edge (one lane per HID channel),
// non-returning relaxed f32 atomics -> L2-resident RMW (agg+vnbr fit in L2).
// ---------------------------------------------------------------------------
__global__ void __launch_bounds__(256)
scatter_kernel(const int* __restrict__ src, const int* __restrict__ dst,
               const float* __restrict__ vnbr, float* __restrict__ agg, int nE) {
    const int t = blockIdx.x * 256 + threadIdx.x;
    const int e = t >> 5;
    if (e >= nE) return;
    const int lane = t & 31;
    const int sn = src[e];
    const int dn = dst[e];
#pragma unroll
    for (int s = 0; s < S_; ++s) {
        const size_t so   = ((size_t)s * N_ + sn) * HID + lane;
        const size_t dofs = ((size_t)s * N_ + dn) * HID + lane;
        const float val = vnbr[so];
        __hip_atomic_fetch_add(&agg[dofs], val, __ATOMIC_RELAXED,
                               __HIP_MEMORY_SCOPE_AGENT);
    }
}

// ---------------------------------------------------------------------------
// Kernel 3: output layer + projection + gate MLP + dh.  One thread per (s,n).
// Weight accesses are uniform -> scalar loads, broadcast from K$.
// ---------------------------------------------------------------------------
__global__ void __launch_bounds__(256)
out_kernel(const float* __restrict__ z, const float* __restrict__ vcur,
           const float* __restrict__ agg,
           const float* __restrict__ outW, const float* __restrict__ outb,
           const float* __restrict__ GW1, const float* __restrict__ Gb1,
           const float* __restrict__ GW2, const float* __restrict__ Gb2,
           float* __restrict__ out, int total) {
    const int idx = blockIdx.x * 256 + threadIdx.x;
    if (idx >= total) return;

    const float* zp = z + (size_t)idx * DZ;
    float c[DIMC], h[DIMH];
#pragma unroll
    for (int i = 0; i < DIMC; ++i) c[i] = zp[i];
#pragma unroll
    for (int i = 0; i < DIMH; ++i) h[i] = zp[DIMC + i];

    float x[2 * HID];
#pragma unroll
    for (int k = 0; k < HID; ++k) {
        x[k]       = vcur[(size_t)idx * HID + k];
        x[HID + k] = agg[(size_t)idx * HID + k];
    }

    // dc = tanh([vcur, agg] @ out_W + out_b)
    float dc[DIMC];
#pragma unroll
    for (int o = 0; o < DIMC; ++o) {
        float acc = outb[o];
#pragma unroll
        for (int k = 0; k < 2 * HID; ++k) acc = fmaf(x[k], outW[k * DIMC + o], acc);
        dc[o] = tanh_fast(acc);
    }

    // project out the component along c
    float s1 = 0.0f, s2 = 0.0f;
#pragma unroll
    for (int i = 0; i < DIMC; ++i) { s1 += dc[i] * c[i]; s2 += c[i] * c[i]; }
    const float r = s1 / s2;
#pragma unroll
    for (int i = 0; i < DIMC; ++i) dc[i] -= r * c[i];

    // G = softplus(mlp(c)); dh = -G * h
    float mid[HID];
#pragma unroll
    for (int j = 0; j < HID; ++j) {
        float acc = Gb1[j];
#pragma unroll
        for (int i = 0; i < DIMC; ++i) acc = fmaf(c[i], GW1[i * HID + j], acc);
        mid[j] = celu1(acc);
    }
    float* op = out + (size_t)idx * DZ;
#pragma unroll
    for (int o = 0; o < DIMH; ++o) {
        float acc = Gb2[o];
#pragma unroll
        for (int j = 0; j < HID; ++j) acc = fmaf(mid[j], GW2[j * DIMH + o], acc);
        op[DIMC + o] = -softplus1(acc) * h[o];
    }
#pragma unroll
    for (int i = 0; i < DIMC; ++i) op[i] = dc[i];
}

// ---------------------------------------------------------------------------
// Host entry
// ---------------------------------------------------------------------------
extern "C" void kernel_launch(void* const* d_in, const int* in_sizes, int n_in,
                              void* d_out, int out_size, void* d_ws, size_t ws_size,
                              hipStream_t stream) {
    (void)in_sizes; (void)n_in; (void)out_size; (void)ws_size;

    const float* z    = (const float*)d_in[1];
    const int*   src  = (const int*)d_in[2];
    const int*   dst  = (const int*)d_in[3];
    const float* cW1  = (const float*)d_in[4];
    const float* cb1  = (const float*)d_in[5];
    const float* cW2  = (const float*)d_in[6];
    const float* cb2  = (const float*)d_in[7];
    const float* nW1  = (const float*)d_in[8];
    const float* nb1  = (const float*)d_in[9];
    const float* nW2  = (const float*)d_in[10];
    const float* nb2  = (const float*)d_in[11];
    const float* outW = (const float*)d_in[12];
    const float* outb = (const float*)d_in[13];
    const float* GW1  = (const float*)d_in[14];
    const float* Gb1  = (const float*)d_in[15];
    const float* GW2  = (const float*)d_in[16];
    const float* Gb2  = (const float*)d_in[17];

    const size_t nv = (size_t)S_ * N_ * HID;   // 6.4M floats each
    float* vcur = (float*)d_ws;
    float* vnbr = vcur + nv;
    float* agg  = vnbr + nv;

    hipMemsetAsync(agg, 0, nv * sizeof(float), stream);

    const int ntiles = (S_ * N_) / 16;                 // 12500 (exact)
    mlp_kernel<<<MLP_BLOCKS, 256, 0, stream>>>(
        z, cW1, cb1, cW2, cb2, nW1, nb1, nW2, nb2, vcur, vnbr, ntiles);

    const long long sthreads = (long long)E_ * 32;
    scatter_kernel<<<(int)((sthreads + 255) / 256), 256, 0, stream>>>(
        src, dst, vnbr, agg, E_);

    const int total = S_ * N_;
    out_kernel<<<(total + 255) / 256, 256, 0, stream>>>(
        z, vcur, agg, outW, outb, GW1, Gb1, GW2, Gb2, (float*)d_out, total);
}
